// Radar4DEncoderPN2_46299747450896
// MI455X (gfx1250) — matrix operand order repaired and checked by reading, not compiled, not run on hardware
//
#include <hip/hip_runtime.h>
#include <hip/hip_bf16.h>
#include <math.h>

// ---------------- CDNA5 WMMA types ----------------
typedef __attribute__((ext_vector_type(16))) _Float16 v16h;
typedef __attribute__((ext_vector_type(8)))  _Float16 v8h;
typedef __attribute__((ext_vector_type(8)))  float    v8f;

#define BB 8          // batch
#define NN0 16384     // input points
#define BN_SCALE 0.9999950000374997f   // 1/sqrt(1+1e-5)

// =====================================================================
// Weight pre-pack: W (Nc,K) f32 row-major -> Wp (Nc,Kp) f16, zero pad.
// =====================================================================
__global__ void pack_w_f16(const float* __restrict__ W, _Float16* __restrict__ Wp,
                           int Nc, int K, int Kp) {
  int t = blockIdx.x * blockDim.x + threadIdx.x;
  if (t >= Nc * Kp) return;
  int n = t / Kp, k = t % Kp;
  Wp[t] = (_Float16)((k < K) ? W[(size_t)n * K + k] : 0.0f);
}

// =====================================================================
// Y[M,Nc] = relu((X[M,KP] @ Wp[Nc,KP]^T) * gamma/sqrt(1+eps) + beta)
// One wave computes a 16 x (16*NT) output strip: A fragment loaded and
// converted once per K-step, reused across NT independent back-to-back
// v_wmma issues (hides XDL latency). M, Nc multiples of 16 by
// construction; X rows zero-padded to KP -> zero conditionals.
// A layout (16-bit 16x32): lane&15 = row; lane<16 -> K {0..7,16..23},
// lane>=16 -> K {8..15,24..31}. B layout (32x16): lane&15 = col;
// lane<16 -> K {0..7,16..23}, lane>=16 -> K {8..15,24..31}.
// =====================================================================
template <int KP, int NT>
__global__ void wmma_linear_bn_relu(const float* __restrict__ X,
                                    const _Float16* __restrict__ Wp,
                                    const float* __restrict__ gam,
                                    const float* __restrict__ bet,
                                    float* __restrict__ Y, int ldy) {
  const int lane = threadIdx.x & 31;
  const int lr   = lane & 15;
  const int half = lane >> 4;
  const int m  = blockIdx.x * 16 + lr;
  const int n0 = blockIdx.y * (16 * NT);

  const float* Xr = X + (size_t)m * KP + half * 8;

  v8f acc[NT];
#pragma unroll
  for (int t = 0; t < NT; ++t) acc[t] = (v8f){};

#pragma unroll
  for (int kk = 0; kk < KP; kk += 32) {
    float4 f0 = *(const float4*)(Xr + kk);
    float4 f1 = *(const float4*)(Xr + kk + 4);
    float4 f2 = *(const float4*)(Xr + kk + 16);
    float4 f3 = *(const float4*)(Xr + kk + 20);
    v16h a;
    a[0]  = (_Float16)f0.x; a[1]  = (_Float16)f0.y;
    a[2]  = (_Float16)f0.z; a[3]  = (_Float16)f0.w;
    a[4]  = (_Float16)f1.x; a[5]  = (_Float16)f1.y;
    a[6]  = (_Float16)f1.z; a[7]  = (_Float16)f1.w;
    a[8]  = (_Float16)f2.x; a[9]  = (_Float16)f2.y;
    a[10] = (_Float16)f2.z; a[11] = (_Float16)f2.w;
    a[12] = (_Float16)f3.x; a[13] = (_Float16)f3.y;
    a[14] = (_Float16)f3.z; a[15] = (_Float16)f3.w;
#pragma unroll
    for (int t = 0; t < NT; ++t) {
      const _Float16* Wr = Wp + (size_t)(n0 + t * 16 + lr) * KP + half * 8;
      v8h b0 = *(const v8h*)(Wr + kk);
      v8h b1 = *(const v8h*)(Wr + kk + 16);
      v16h b;
#pragma unroll
      for (int j = 0; j < 8; ++j) { b[j] = b0[j]; b[j + 8] = b1[j]; }
      acc[t] = __builtin_amdgcn_wmma_f32_16x16x32_f16(false, a, false, b,
                                                      (short)0, acc[t],
                                                      false, false);
    }
  }

#pragma unroll
  for (int t = 0; t < NT; ++t) {
    int n = n0 + t * 16 + lr;
    float s  = gam[n] * BN_SCALE;
    float bb = bet[n];
    float* yr = Y + (size_t)(blockIdx.x * 16 + half * 8) * ldy + n;
#pragma unroll
    for (int j = 0; j < 8; ++j) {
      float v = acc[t][j] * s + bb;
      yr[(size_t)j * ldy] = fmaxf(v, 0.0f);
    }
  }
}

// =====================================================================
// Farthest point sampling: one block per batch, exact argmax tie-break.
// =====================================================================
__global__ void fps_kernel(const float* __restrict__ xyz, float* __restrict__ dmin,
                           int* __restrict__ fidx, int N, int np) {
  int b = blockIdx.x, tid = threadIdx.x;
  const float* xb = xyz + (size_t)b * N * 3;
  float* dm = dmin + (size_t)b * N;
  __shared__ float sval[256];
  __shared__ int   sidx[256];
  __shared__ int   slast;
  for (int i = tid; i < N; i += 256) dm[i] = 1e10f;
  if (tid == 0) { fidx[b * np] = 0; slast = 0; }
  __syncthreads();
  for (int step = 1; step < np; ++step) {
    int last = slast;
    float px = xb[last * 3 + 0], py = xb[last * 3 + 1], pz = xb[last * 3 + 2];
    float bv = -1.0f; int bi = 0;
    for (int i = tid; i < N; i += 256) {
      float dx = xb[i * 3 + 0] - px, dy = xb[i * 3 + 1] - py, dz = xb[i * 3 + 2] - pz;
      float d = dx * dx + dy * dy + dz * dz;
      float v = dm[i]; v = fminf(v, d); dm[i] = v;
      if (v > bv) { bv = v; bi = i; }
    }
    sval[tid] = bv; sidx[tid] = bi;
    __syncthreads();
    for (int off = 128; off > 0; off >>= 1) {
      if (tid < off) {
        if (sval[tid + off] > sval[tid] ||
            (sval[tid + off] == sval[tid] && sidx[tid + off] < sidx[tid])) {
          sval[tid] = sval[tid + off]; sidx[tid] = sidx[tid + off];
        }
      }
      __syncthreads();
    }
    if (tid == 0) { fidx[b * np + step] = sidx[0]; slast = sidx[0]; }
    __syncthreads();
  }
}

__global__ void gather_xyz(const float* __restrict__ xyz, const int* __restrict__ fidx,
                           float* __restrict__ out, int N, int S) {
  int t = blockIdx.x * blockDim.x + threadIdx.x;
  if (t >= BB * S) return;
  int b = t / S;
  int id = fidx[t];
  const float* p = xyz + ((size_t)b * N + id) * 3;
  out[t * 3 + 0] = p[0]; out[t * 3 + 1] = p[1]; out[t * 3 + 2] = p[2];
}

// =====================================================================
// Ball query: one wave per center; ordered first-k-within-radius via
// wave32 ballot; pad with first found index (0 if none) per reference.
// =====================================================================
__global__ void ball_query(const float* __restrict__ src, const float* __restrict__ ctr,
                           int* __restrict__ idxo, int Nn, int S, int k, float r2) {
  int lane = threadIdx.x & 31;
  int wid = (blockIdx.x * blockDim.x + threadIdx.x) >> 5;
  if (wid >= BB * S) return;
  int b = wid / S;
  float cx = ctr[wid * 3 + 0], cy = ctr[wid * 3 + 1], cz = ctr[wid * 3 + 2];
  const float* sb = src + (size_t)b * Nn * 3;
  int* out = idxo + (size_t)wid * k;
  int filled = 0, first = 0;
  bool have = false;
  for (int base = 0; base < Nn && filled < k; base += 32) {
    int p = base + lane;
    bool in = false;
    if (p < Nn) {
      float dx = sb[p * 3 + 0] - cx, dy = sb[p * 3 + 1] - cy, dz = sb[p * 3 + 2] - cz;
      in = (dx * dx + dy * dy + dz * dz) < r2;
    }
    unsigned mask = (unsigned)__ballot(in);
    if (mask && !have) { first = base + __builtin_ctz(mask); have = true; }
    if (in) {
      int pos = filled + __popc(mask & ((1u << lane) - 1u));
      if (pos < k) out[pos] = p;
    }
    filled += __popc(mask);
  }
  int f = filled < k ? filled : k;
  for (int pos = f + lane; pos < k; pos += 32) out[pos] = first;
}

// Grouped SA input: [rel(3), feat(Ci), zeros...] rows of width ldp.
__global__ void build_sa(const float* __restrict__ sx, const float* __restrict__ sf,
                         const float* __restrict__ ctr, const int* __restrict__ idx,
                         float* __restrict__ Xg, int Nn, int S, int k, int Ci, int ldp) {
  int row = blockIdx.x * blockDim.x + threadIdx.x;
  if (row >= BB * S * k) return;
  int w = row / k;
  int b = w / S;
  int id = idx[row];
  float* o = Xg + (size_t)row * ldp;
  const float* pv = sx + ((size_t)b * Nn + id) * 3;
  o[0] = pv[0] - ctr[w * 3 + 0];
  o[1] = pv[1] - ctr[w * 3 + 1];
  o[2] = pv[2] - ctr[w * 3 + 2];
  const float* fv = sf + ((size_t)b * Nn + id) * Ci;
  for (int c = 0; c < Ci; ++c) o[3 + c] = fv[c];
  for (int c = 3 + Ci; c < ldp; ++c) o[c] = 0.0f;
}

// Cost-volume input: [ft_center(64), ft1[idx](64), rel(3), zeros...] width 160.
__global__ void build_cv(const float* __restrict__ sx, const float* __restrict__ sf,
                         const float* __restrict__ cx_, const float* __restrict__ cf,
                         const int* __restrict__ idx, float* __restrict__ Xg,
                         int S, int k) {
  int row = blockIdx.x * blockDim.x + threadIdx.x;
  if (row >= BB * S * k) return;
  int w = row / k;
  int b = w / S;
  int id = idx[row];
  float* o = Xg + (size_t)row * 160;
  const float* fc = cf + (size_t)w * 64;
  for (int c = 0; c < 64; ++c) o[c] = fc[c];
  const float* f1 = sf + ((size_t)b * S + id) * 64;
  for (int c = 0; c < 64; ++c) o[64 + c] = f1[c];
  const float* pv = sx + ((size_t)b * S + id) * 3;
  o[128] = pv[0] - cx_[w * 3 + 0];
  o[129] = pv[1] - cx_[w * 3 + 1];
  o[130] = pv[2] - cx_[w * 3 + 2];
  for (int c = 131; c < 160; ++c) o[c] = 0.0f;
}

__global__ void maxpool_k(const float* __restrict__ H, float* __restrict__ out,
                          int S, int k, int ldo, int coff) {
  int t = blockIdx.x * blockDim.x + threadIdx.x;
  if (t >= BB * S * 64) return;
  int ch = t % 64, w = t / 64;
  const float* h = H + (size_t)w * k * 64 + ch;
  float m = h[0];
  for (int j = 1; j < k; ++j) m = fmaxf(m, h[(size_t)j * 64]);
  out[(size_t)w * ldo + coff + ch] = m;
}

__global__ void copycols64(const float* __restrict__ src, float* __restrict__ dst,
                           int rows, int ldd, int coff) {
  int t = blockIdx.x * blockDim.x + threadIdx.x;
  if (t >= rows * 64) return;
  int c = t % 64, r = t / 64;
  dst[(size_t)r * ldd + coff + c] = src[(size_t)r * 64 + c];
}

// Exact 3-NN (strict-less insertion matches top_k tie order) + IDW weights.
__global__ void knn3(const float* __restrict__ sx, const float* __restrict__ dx_,
                     int Ss, int Sd, int* __restrict__ ki, float* __restrict__ kw) {
  int t = blockIdx.x * blockDim.x + threadIdx.x;
  if (t >= BB * Sd) return;
  int b = t / Sd;
  float px = dx_[t * 3 + 0], py = dx_[t * 3 + 1], pz = dx_[t * 3 + 2];
  const float* sb = sx + (size_t)b * Ss * 3;
  float bd0 = 1e30f, bd1 = 1e30f, bd2 = 1e30f;
  int bi0 = 0, bi1 = 0, bi2 = 0;
  for (int j = 0; j < Ss; ++j) {
    float dx = sb[j * 3 + 0] - px, dy = sb[j * 3 + 1] - py, dz = sb[j * 3 + 2] - pz;
    float d = dx * dx + dy * dy + dz * dz;
    if (d < bd0)      { bd2 = bd1; bi2 = bi1; bd1 = bd0; bi1 = bi0; bd0 = d; bi0 = j; }
    else if (d < bd1) { bd2 = bd1; bi2 = bi1; bd1 = d; bi1 = j; }
    else if (d < bd2) { bd2 = d; bi2 = j; }
  }
  float w0 = 1.0f / (bd0 + 1e-8f);
  float w1 = 1.0f / (bd1 + 1e-8f);
  float w2 = 1.0f / (bd2 + 1e-8f);
  float ws = w0 + w1 + w2;
  kw[t * 3 + 0] = w0 / ws; kw[t * 3 + 1] = w1 / ws; kw[t * 3 + 2] = w2 / ws;
  ki[t * 3 + 0] = bi0; ki[t * 3 + 1] = bi1; ki[t * 3 + 2] = bi2;
}

__global__ void interp_apply(const float* __restrict__ sf, const int* __restrict__ ki,
                             const float* __restrict__ kw, float* __restrict__ df,
                             int Ss, int Sd) {
  int t = blockIdx.x * blockDim.x + threadIdx.x;
  if (t >= BB * Sd * 64) return;
  int ch = t % 64, w = t / 64;
  int b = w / Sd;
  const float* sfb = sf + (size_t)b * Ss * 64;
  float acc = 0.0f;
  for (int j = 0; j < 3; ++j)
    acc += kw[w * 3 + j] * sfb[(size_t)ki[w * 3 + j] * 64 + ch];
  df[(size_t)w * 64 + ch] = acc;
}

// =====================================================================
// Host orchestration
// =====================================================================
extern "C" void kernel_launch(void* const* d_in, const int* in_sizes, int n_in,
                              void* d_out, int out_size, void* d_ws, size_t ws_size,
                              hipStream_t stream) {
  if (n_in < 199) return;
  const float* Pp[199];
  for (int i = 0; i < 199; ++i) Pp[i] = (const float*)d_in[i];

  // ---- workspace carve-up (base is 256B-aligned; keep chunks aligned) ----
  char* wsb = (char*)d_ws;
  size_t off = 0;
  auto alloc = [&](size_t bytes) -> void* {
    void* p = wsb + off;
    off = (off + bytes + 255) & ~(size_t)255;
    return p;
  };
  float* dmin = (float*)alloc((size_t)BB * NN0 * 4);
  int*   fidx = (int*)alloc((size_t)BB * 256 * 4);
  int*   idx  = (int*)alloc((size_t)BB * 256 * 32 * 4);
  float* Xg   = (float*)alloc((size_t)65536 * 160 * 4);
  float* H1   = (float*)alloc((size_t)65536 * 32 * 4);
  float* H2   = (float*)alloc((size_t)65536 * 64 * 4);
  float* bcc  = (float*)alloc((size_t)BB * 256 * 128 * 4);
  float* pcc  = (float*)alloc((size_t)BB * 256 * 192 * 4);
  float* xt[2][4]; float* ft[2][4]; float* cvb[4];
  for (int f = 0; f < 2; ++f)
    for (int l = 0; l < 4; ++l) {
      xt[f][l] = (float*)alloc((size_t)BB * 256 * 3 * 4);
      ft[f][l] = (float*)alloc((size_t)BB * 256 * 64 * 4);
    }
  for (int l = 0; l < 4; ++l) cvb[l] = (float*)alloc((size_t)BB * 256 * 64 * 4);
  float* xA = (float*)alloc((size_t)BB * 256 * 64 * 4);
  float* xB = (float*)alloc((size_t)BB * 256 * 64 * 4);
  float* xu = (float*)alloc((size_t)BB * 256 * 64 * 4);
  int*   ki = (int*)alloc((size_t)BB * 256 * 3 * 4);
  float* kw = (float*)alloc((size_t)BB * 256 * 3 * 4);
  _Float16* wpool = (_Float16*)alloc((size_t)2 * 1024 * 1024);  // packed weights
  size_t wcur = 0;

  auto g1 = [](int n) { return dim3((unsigned)((n + 255) / 256)); };

  // Pack weights + launch templated WMMA linear. X row stride must be Kp.
  auto linear = [&](const float* X, int M, int K, int Nc,
                    const float* Wraw, const float* gm, const float* bt,
                    float* Y, int ldy) {
    int Kp = (K + 31) & ~31;
    int NT = (Nc == 32) ? 2 : 4;    // Nc in {32,64,256}; Nc % (16*NT) == 0
    _Float16* wp = wpool + wcur;
    wcur += (size_t)Nc * Kp;
    pack_w_f16<<<g1(Nc * Kp), 256, 0, stream>>>(Wraw, wp, Nc, K, Kp);
    dim3 grid((unsigned)(M / 16), (unsigned)(Nc / (16 * NT)));
#define WMMA_CASE(KPv, NTv)                                                   \
    if (Kp == KPv && NT == NTv) {                                             \
      wmma_linear_bn_relu<KPv, NTv><<<grid, 32, 0, stream>>>(X, wp, gm, bt,   \
                                                             Y, ldy);         \
      return;                                                                 \
    }
    WMMA_CASE(32, 2)  WMMA_CASE(96, 2)  WMMA_CASE(160, 2)
    WMMA_CASE(32, 4)  WMMA_CASE(64, 4)  WMMA_CASE(128, 4) WMMA_CASE(192, 4)
    WMMA_CASE(64, 2)  WMMA_CASE(128, 2) WMMA_CASE(192, 2)
    WMMA_CASE(96, 4)  WMMA_CASE(160, 4)
#undef WMMA_CASE
  };
  auto mlp2 = [&](const float* X, int M, int K1, int hid, int Nout,
                  int pbase, float* Hbuf, float* Y, int ldy) {
    linear(X, M, K1, hid, Pp[pbase], Pp[pbase + 1], Pp[pbase + 2], Hbuf, hid);
    linear(Hbuf, M, hid, Nout, Pp[pbase + 3], Pp[pbase + 4], Pp[pbase + 5], Y, ldy);
  };

  // Param index map (insertion-order flattening):
  // 0..3 inputs; sa_t: 4+lvl*18; sa_t1: 76+lvl*18; cv: 148+i*6;
  // proj4:172 proj3:178 proj2:184 proj1:190 out:196
  auto sa_base = [](int f, int lvl) { return 4 + (f * 4 + lvl) * 18; };

  const float* xyz_in[2]  = {(const float*)d_in[0], (const float*)d_in[2]};
  const float* feat_in[2] = {(const float*)d_in[1], (const float*)d_in[3]};
  const int NP[4] = {256, 128, 64, 32};

  // ---- pyramids for both frames ----
  for (int f = 0; f < 2; ++f) {
    const float* cx = xyz_in[f];
    const float* cf = feat_in[f];
    int curN = NN0, curC = 3;
    for (int lvl = 0; lvl < 4; ++lvl) {
      int S = NP[lvl];
      fps_kernel<<<BB, 256, 0, stream>>>(cx, dmin, fidx, curN, S);
      gather_xyz<<<g1(BB * S), 256, 0, stream>>>(cx, fidx, xt[f][lvl], curN, S);
      int base = sa_base(f, lvl);
      int Kin = 3 + curC;
      int ldp = (Kin + 31) & ~31;   // padded X row stride == Kp of layer 1
      for (int br = 0; br < 2; ++br) {
        int k = (br == 0) ? 16 : 32;
        float r = 0.05f * (float)(1 << (lvl + br));
        ball_query<<<g1(BB * S * 32), 256, 0, stream>>>(cx, xt[f][lvl], idx,
                                                        curN, S, k, r * r);
        int rows = BB * S * k;
        build_sa<<<g1(rows), 256, 0, stream>>>(cx, cf, xt[f][lvl], idx, Xg,
                                               curN, S, k, curC, ldp);
        mlp2(Xg, rows, Kin, 32, 64, base + br * 6, H1, H2, 64);
        maxpool_k<<<g1(BB * S * 64), 256, 0, stream>>>(H2, bcc, S, k, 128, br * 64);
      }
      mlp2(bcc, BB * S, 128, 64, 64, base + 12, H2, ft[f][lvl], 64);
      cx = xt[f][lvl]; cf = ft[f][lvl]; curN = S; curC = 64;
    }
  }

  // ---- cost volumes ----
  for (int i = 0; i < 4; ++i) {
    int S = NP[i];
    float r = 0.05f * (float)(1 << (i + 1));  // CV_R[i]
    ball_query<<<g1(BB * S * 32), 256, 0, stream>>>(xt[1][i], xt[0][i], idx,
                                                    S, S, 32, r * r);
    int rows = BB * S * 32;
    build_cv<<<g1(rows), 256, 0, stream>>>(xt[1][i], ft[1][i], xt[0][i], ft[0][i],
                                           idx, Xg, S, 32);
    mlp2(Xg, rows, 131, 32, 64, 148 + i * 6, H1, H2, 64);
    maxpool_k<<<g1(BB * S * 64), 256, 0, stream>>>(H2, cvb[i], S, 32, 64, 0);
  }

  // ---- decoder ----
  {  // proj4 on [ft3, cv3] (rows = 8*32, K=128) -> x4 (xA)
    int rows = BB * 32;
    copycols64<<<g1(rows * 64), 256, 0, stream>>>(ft[0][3], pcc, rows, 128, 0);
    copycols64<<<g1(rows * 64), 256, 0, stream>>>(cvb[3], pcc, rows, 128, 64);
    mlp2(pcc, rows, 128, 64, 64, 172, H2, xA, 64);
  }
  {  // interp 32 -> 64, proj3 -> x3 (xB)
    knn3<<<g1(BB * 64), 256, 0, stream>>>(xt[0][3], xt[0][2], 32, 64, ki, kw);
    interp_apply<<<g1(BB * 64 * 64), 256, 0, stream>>>(xA, ki, kw, xu, 32, 64);
    int rows = BB * 64;
    copycols64<<<g1(rows * 64), 256, 0, stream>>>(ft[0][2], pcc, rows, 192, 0);
    copycols64<<<g1(rows * 64), 256, 0, stream>>>(cvb[2], pcc, rows, 192, 64);
    copycols64<<<g1(rows * 64), 256, 0, stream>>>(xu, pcc, rows, 192, 128);
    mlp2(pcc, rows, 192, 64, 64, 178, H2, xB, 64);
  }
  {  // interp 64 -> 128, proj2 -> x2 (xA)
    knn3<<<g1(BB * 128), 256, 0, stream>>>(xt[0][2], xt[0][1], 64, 128, ki, kw);
    interp_apply<<<g1(BB * 128 * 64), 256, 0, stream>>>(xB, ki, kw, xu, 64, 128);
    int rows = BB * 128;
    copycols64<<<g1(rows * 64), 256, 0, stream>>>(ft[0][1], pcc, rows, 192, 0);
    copycols64<<<g1(rows * 64), 256, 0, stream>>>(cvb[1], pcc, rows, 192, 64);
    copycols64<<<g1(rows * 64), 256, 0, stream>>>(xu, pcc, rows, 192, 128);
    mlp2(pcc, rows, 192, 64, 64, 184, H2, xA, 64);
  }
  {  // interp 128 -> 256, proj1 -> x1 (xB)
    knn3<<<g1(BB * 256), 256, 0, stream>>>(xt[0][1], xt[0][0], 128, 256, ki, kw);
    interp_apply<<<g1(BB * 256 * 64), 256, 0, stream>>>(xA, ki, kw, xu, 128, 256);
    int rows = BB * 256;
    copycols64<<<g1(rows * 64), 256, 0, stream>>>(ft[0][0], pcc, rows, 192, 0);
    copycols64<<<g1(rows * 64), 256, 0, stream>>>(cvb[0], pcc, rows, 192, 64);
    copycols64<<<g1(rows * 64), 256, 0, stream>>>(xu, pcc, rows, 192, 128);
    mlp2(pcc, rows, 192, 64, 64, 190, H2, xB, 64);
  }
  // out layer: (8*256, 64) -> (8*256, 256) into d_out (relu+BN per mlp_apply)
  linear(xB, BB * 256, 64, 256, Pp[196], Pp[197], Pp[198], (float*)d_out, 256);
}
